// MyEncoder_52621939310797
// MI455X (gfx1250) — matrix operand order repaired and checked by reading, not compiled, not run on hardware
//
#include <hip/hip_runtime.h>

#define T_ 256
#define B_ 64
#define H_ 1024
#define G_ 4096   // 4*H
#define L_ 2
#define KC_ (H_ / 32)        // 32 K-chunks per 1024-deep reduction
#define TILE_ (32 * 16)      // 512 bf16 elems per packed 16x32 fragment tile
#define NTBLK_ (KC_ * TILE_) // 16384 elems per (16-col x full-K) packed block

typedef __bf16 bf16_t;
typedef __attribute__((ext_vector_type(16))) __bf16 v16bf;
typedef __attribute__((ext_vector_type(8)))  __bf16 v8bf;
typedef __attribute__((ext_vector_type(8)))  float  v8f;

// ---------------------------------------------------------------- helpers
__device__ __forceinline__ v8f wmma_bf(v16bf a, v16bf b, v8f c) {
  return __builtin_amdgcn_wmma_f32_16x16x32_bf16(
      false, a, false, b, (short)0, c, false, false);
}

// Packed fragment: this lane's 16 bf16 values are contiguous (32 B).
__device__ __forceinline__ v16bf ld32(const bf16_t* p) {
  v8bf lo = *(const v8bf*)(p);
  v8bf hi = *(const v8bf*)(p + 8);
  return __builtin_shufflevector(lo, hi, 0,1,2,3,4,5,6,7,8,9,10,11,12,13,14,15);
}

// Position of element (row, j) of a 64x1024 activation matrix in packed
// WMMA-A fragment order. A-layout (16-bit, 16x32): lane = (row&15) + 16*lhi,
// lane holds K = lhi*8+{0..7} and lhi*8+16+{0..7} (elems 0..7 / 8..15).
__device__ __forceinline__ size_t pack_idx(int row, int j) {
  int mt = row >> 4, r = row & 15;
  int kcidx = j >> 5, kk = j & 31;
  int lhi = (kk >> 3) & 1;
  int e = (kk & 7) + ((kk >> 4) << 3);
  int lane = r + (lhi << 4);
  return ((size_t)(mt * KC_ + kcidx) * 32 + lane) * 16 + e;
}

// ---------------------------------------------------------------- kernels
__global__ void k_embed(const int* __restrict__ tokens,
                        const float* __restrict__ emb,
                        bf16_t* __restrict__ xbf) {
  size_t tid = (size_t)blockIdx.x * blockDim.x + threadIdx.x;   // T*B*H
  size_t tb = tid >> 10;            // t*B + b
  int j = (int)(tid & (H_ - 1));
  int t = (int)(tb >> 6);
  int b = (int)(tb & (B_ - 1));
  int tok = tokens[tb];
  float v = emb[(size_t)tok * H_ + j];
  xbf[(size_t)t * (B_ * H_) + pack_idx(b, j)] = (bf16_t)v;
}

// Pack weight matrix (rows = output gate g, cols = input k) into WMMA-B
// fragment order. B-layout (32x16): lane = (col&15) + 16*lhi, lane holds
// K = lhi*8+{0..7} and lhi*8+16+{0..7} (elems 0..7 / 8..15).
__global__ void k_pack_w(const float* __restrict__ W, bf16_t* __restrict__ P,
                         int n) {
  int p = blockIdx.x * blockDim.x + threadIdx.x;
  if (p >= n) return;
  int e     = p & 15;
  int lane  = (p >> 4) & 31;
  int kcidx = (p >> 9) & (KC_ - 1);
  int nt    = p >> 14;                     // 16-col block (spans both layers)
  int col   = nt * 16 + (lane & 15);
  int k     = kcidx * 32 + ((lane >> 4) << 3) + (e & 7) + ((e >> 3) << 4);
  P[p] = (bf16_t)W[(size_t)col * H_ + k];
}

__global__ void k_zero_state(float* __restrict__ h, float* __restrict__ c,
                             bf16_t* __restrict__ hbf) {
  int i = blockIdx.x * blockDim.x + threadIdx.x;  // B*H
  h[i] = 0.f; c[i] = 0.f; hbf[i] = (bf16_t)0.f;
}

// pre[b,g] = sum_k x[t,b,k]*Wi[g,k] + sum_k h[b,k]*Wh[g,k]
// 32 blocks x 256 threads = 256 waves; wave owns a 16x64 strip (4 accums).
// All fragment loads are coalesced (lane*32B) and double-buffered.
__global__ __launch_bounds__(256)
void k_lstm_gemm(const bf16_t* __restrict__ xbf,   // packed [T][4mt][KC][32][16]
                 const bf16_t* __restrict__ hbf,   // packed [4mt][KC][32][16]
                 const bf16_t* __restrict__ Wi,    // packed [G/16][KC][32][16]
                 const bf16_t* __restrict__ Wh,    // packed
                 float* __restrict__ pre,          // [B,G]
                 int t) {
  const int lane = threadIdx.x & 31;
  const int wave = threadIdx.x >> 5;
  const int job  = blockIdx.x * 8 + wave;   // 0..255
  const int mtile = job >> 6;               // 0..3
  const int ngrp  = job & 63;               // 0..63
  const int nt0   = ngrp * 4;

  const bf16_t* pax = xbf + (size_t)t * (B_ * H_) +
                      (size_t)mtile * NTBLK_ + lane * 16;
  const bf16_t* pah = hbf + (size_t)mtile * NTBLK_ + lane * 16;
  const bf16_t* pi0 = Wi + (size_t)(nt0 + 0) * NTBLK_ + lane * 16;
  const bf16_t* pi1 = Wi + (size_t)(nt0 + 1) * NTBLK_ + lane * 16;
  const bf16_t* pi2 = Wi + (size_t)(nt0 + 2) * NTBLK_ + lane * 16;
  const bf16_t* pi3 = Wi + (size_t)(nt0 + 3) * NTBLK_ + lane * 16;
  const bf16_t* ph0 = Wh + (size_t)(nt0 + 0) * NTBLK_ + lane * 16;
  const bf16_t* ph1 = Wh + (size_t)(nt0 + 1) * NTBLK_ + lane * 16;
  const bf16_t* ph2 = Wh + (size_t)(nt0 + 2) * NTBLK_ + lane * 16;
  const bf16_t* ph3 = Wh + (size_t)(nt0 + 3) * NTBLK_ + lane * 16;

  v8f a0 = {}, a1 = {}, a2 = {}, a3 = {};

  // ---- x * Wi^T, software pipelined ----
  {
    v16bf A  = ld32(pax);
    v16bf b0 = ld32(pi0), b1 = ld32(pi1), b2 = ld32(pi2), b3 = ld32(pi3);
#pragma unroll 2
    for (int kc = 1; kc < KC_; ++kc) {
      const int off = kc * TILE_;
      v16bf An  = ld32(pax + off);
      v16bf b0n = ld32(pi0 + off);
      v16bf b1n = ld32(pi1 + off);
      v16bf b2n = ld32(pi2 + off);
      v16bf b3n = ld32(pi3 + off);
      a0 = wmma_bf(A, b0, a0);
      a1 = wmma_bf(A, b1, a1);
      a2 = wmma_bf(A, b2, a2);
      a3 = wmma_bf(A, b3, a3);
      A = An; b0 = b0n; b1 = b1n; b2 = b2n; b3 = b3n;
    }
    a0 = wmma_bf(A, b0, a0);
    a1 = wmma_bf(A, b1, a1);
    a2 = wmma_bf(A, b2, a2);
    a3 = wmma_bf(A, b3, a3);
  }
  // ---- h * Wh^T, software pipelined ----
  {
    v16bf A  = ld32(pah);
    v16bf b0 = ld32(ph0), b1 = ld32(ph1), b2 = ld32(ph2), b3 = ld32(ph3);
#pragma unroll 2
    for (int kc = 1; kc < KC_; ++kc) {
      const int off = kc * TILE_;
      v16bf An  = ld32(pah + off);
      v16bf b0n = ld32(ph0 + off);
      v16bf b1n = ld32(ph1 + off);
      v16bf b2n = ld32(ph2 + off);
      v16bf b3n = ld32(ph3 + off);
      a0 = wmma_bf(A, b0, a0);
      a1 = wmma_bf(A, b1, a1);
      a2 = wmma_bf(A, b2, a2);
      a3 = wmma_bf(A, b3, a3);
      A = An; b0 = b0n; b1 = b1n; b2 = b2n; b3 = b3n;
    }
    a0 = wmma_bf(A, b0, a0);
    a1 = wmma_bf(A, b1, a1);
    a2 = wmma_bf(A, b2, a2);
    a3 = wmma_bf(A, b3, a3);
  }

  // C/D layout: lane holds col lane&15, rows r + 8*(lane>>4)
  const int lrow = lane & 15;
  const int mrow0 = mtile * 16 + (lane >> 4) * 8;
  const int nbase = ngrp * 64;
#pragma unroll
  for (int r = 0; r < 8; ++r) {
    size_t off = (size_t)(mrow0 + r) * G_ + nbase + lrow;
    pre[off +  0] = a0[r];
    pre[off + 16] = a1[r];
    pre[off + 32] = a2[r];
    pre[off + 48] = a3[r];
  }
}

__global__ void k_gates(const float* __restrict__ pre,
                        const float* __restrict__ bi,
                        const float* __restrict__ bhb,
                        const int* __restrict__ lengths,
                        float* __restrict__ h, float* __restrict__ c,
                        bf16_t* __restrict__ hbf,         // packed
                        float* __restrict__ hs_f32,       // nullable, plain
                        bf16_t* __restrict__ hs_bf,       // nullable, packed
                        int t) {
  int tid = blockIdx.x * blockDim.x + threadIdx.x;  // B*H
  int b = tid >> 10;
  int j = tid & (H_ - 1);
  const float* pr = pre + (size_t)b * G_;
  float pf = pr[j]          + bi[j]          + bhb[j];
  float pi = pr[H_ + j]     + bi[H_ + j]     + bhb[H_ + j];
  float po = pr[2 * H_ + j] + bi[2 * H_ + j] + bhb[2 * H_ + j];
  float pg = pr[3 * H_ + j] + bi[3 * H_ + j] + bhb[3 * H_ + j];
  float fg = 1.f / (1.f + __expf(-pf));
  float ig = 1.f / (1.f + __expf(-pi));
  float og = 1.f / (1.f + __expf(-po));
  float gg = tanhf(pg);
  float co = c[tid], ho = h[tid];
  float cn = fg * co + ig * gg;
  float hn = og * tanhf(co);          // reference uses OLD cell here
  bool upd = t < lengths[b];
  float hv = upd ? hn : ho;
  float cv = upd ? cn : co;
  size_t pidx = pack_idx(b, j);
  h[tid] = hv; c[tid] = cv; hbf[pidx] = (bf16_t)hv;
  if (hs_f32) hs_f32[(size_t)t * (B_ * H_) + tid] = hv;
  if (hs_bf)  hs_bf [(size_t)t * (B_ * H_) + pidx] = (bf16_t)hv;
}

__global__ void k_copy_state(const float* __restrict__ h,
                             const float* __restrict__ c,
                             float* __restrict__ out_h,
                             float* __restrict__ out_c) {
  int i = blockIdx.x * blockDim.x + threadIdx.x;  // B*H
  out_h[i] = h[i];
  out_c[i] = c[i];
}

// ---------------------------------------------------------------- launch
extern "C" void kernel_launch(void* const* d_in, const int* in_sizes, int n_in,
                              void* d_out, int out_size, void* d_ws, size_t ws_size,
                              hipStream_t stream) {
  const int*   tokens  = (const int*)d_in[0];    // [T,B]
  const int*   lengths = (const int*)d_in[1];    // [B]
  const float* emb     = (const float*)d_in[2];  // [V,H]
  const float* Wi      = (const float*)d_in[3];  // [L,G,H]
  const float* bi      = (const float*)d_in[4];  // [L,G]
  const float* Wh      = (const float*)d_in[5];  // [L,G,H]
  const float* bh      = (const float*)d_in[6];  // [L,G]
  float* out = (float*)d_out;

  char* ws = (char*)d_ws;
  bf16_t* xbf  = (bf16_t*)ws; ws += (size_t)T_ * B_ * H_ * 2;   // 32 MB, packed
  bf16_t* Wibf = (bf16_t*)ws; ws += (size_t)L_ * G_ * H_ * 2;   // 16 MB, packed
  bf16_t* Whbf = (bf16_t*)ws; ws += (size_t)L_ * G_ * H_ * 2;   // 16 MB, packed
  float*  h    = (float*)ws;  ws += (size_t)B_ * H_ * 4;
  float*  c    = (float*)ws;  ws += (size_t)B_ * H_ * 4;
  bf16_t* hbf  = (bf16_t*)ws; ws += (size_t)B_ * H_ * 2;        // packed
  float*  pre  = (float*)ws;  ws += (size_t)B_ * G_ * 4;        // 1 MB

  k_embed<<<(T_ * B_ * H_) / 256, 256, 0, stream>>>(tokens, emb, xbf);

  const int nW = L_ * G_ * H_;
  k_pack_w<<<(nW + 255) / 256, 256, 0, stream>>>(Wi, Wibf, nW);
  k_pack_w<<<(nW + 255) / 256, 256, 0, stream>>>(Wh, Whbf, nW);

  float* out_h = out;                            // [L,B,H]
  float* out_c = out + (size_t)L_ * B_ * H_;     // [L,B,H]
  float* out_x = out + (size_t)2 * L_ * B_ * H_; // [T,B,H]

  for (int l = 0; l < L_; ++l) {
    k_zero_state<<<(B_ * H_) / 256, 256, 0, stream>>>(h, c, hbf);
    const bf16_t* Wil = Wibf + (size_t)l * G_ * H_;
    const bf16_t* Whl = Whbf + (size_t)l * G_ * H_;
    const float*  bil = bi + (size_t)l * G_;
    const float*  bhl = bh + (size_t)l * G_;
    for (int t = 0; t < T_; ++t) {
      k_lstm_gemm<<<32, 256, 0, stream>>>(xbf, hbf, Wil, Whl, pre, t);
      float*  hsf = (l == L_ - 1) ? out_x : nullptr;
      bf16_t* hsb = (l == L_ - 1) ? nullptr : xbf;  // next layer's input, packed
      k_gates<<<(B_ * H_) / 256, 256, 0, stream>>>(pre, bil, bhl, lengths,
                                                   h, c, hbf, hsf, hsb, t);
    }
    k_copy_state<<<(B_ * H_) / 256, 256, 0, stream>>>(
        h, c, out_h + (size_t)l * B_ * H_, out_c + (size_t)l * B_ * H_);
  }
}
